// PhpNetGraphTokens_62010737820203
// MI455X (gfx1250) — matrix operand order repaired and checked by reading, not compiled, not run on hardware
//
#include <hip/hip_runtime.h>
#include <stdint.h>

#define NN     4096     // nodes (padded, constant through all layers)
#define EE     16384    // edge slots (constant, power of two -> bitonic)
#define NBATCH 8
#define FF     6000     // hidden feature width
#define FIN    4000     // embedding-concat width
#define TOKN   20
#define EMBD   200

#define BM 128
#define BN 128
#define BK 32
#define LDST 40         // LDS row stride in halfs (BK + 8 pad)

typedef _Float16 f16;
typedef __attribute__((ext_vector_type(4)))  _Float16 h4;
typedef __attribute__((ext_vector_type(8)))  _Float16 v8h;
typedef __attribute__((ext_vector_type(16))) _Float16 v16h;
typedef __attribute__((ext_vector_type(8)))  float    v8f;
typedef __attribute__((ext_vector_type(4)))  float    f4;

union frag16 { v16h v; v8h h[2]; };

// ---------------------------------------------------------------- utilities
static __device__ __forceinline__ void atomicMaxF(float* addr, float val) {
  unsigned int* ua = (unsigned int*)addr;
  unsigned int old = *ua;
  while (__uint_as_float(old) < val) {
    unsigned int assumed = old;
    old = atomicCAS(ua, assumed, __float_as_uint(val));
    if (old == assumed) break;
  }
}

__global__ void k_fill_f32(float* p, float v, int n) {
  int i = blockIdx.x * blockDim.x + threadIdx.x;
  if (i < n) p[i] = v;
}
__global__ void k_fill_i32(int* p, int v, int n) {
  int i = blockIdx.x * blockDim.x + threadIdx.x;
  if (i < n) p[i] = v;
}
__global__ void k_copy_i32(int* dst, const int* src, int n) {
  int i = blockIdx.x * blockDim.x + threadIdx.x;
  if (i < n) dst[i] = src[i];
}

// ---------------------------------------------------------------- embedding
__global__ void k_embed(const int* __restrict__ tokens, const float* __restrict__ emb,
                        float* __restrict__ x) {
  int i = blockIdx.x * blockDim.x + threadIdx.x;
  if (i >= NN * FIN) return;
  int v = i / FIN;
  int r = i - v * FIN;
  int t = r / EMBD;
  int e = r - t * EMBD;
  x[i] = emb[(size_t)tokens[v * TOKN + t] * EMBD + e];
}

// ---------------------------------------------------------------- WMMA GEMM
// C[M,Nc] = A[M,K] @ B[K,Nc] (+bias, optional relu). fp32 in HBM (memory
// bound: one streaming pass of the weights), f16 WMMA math, f32 accumulate.
// 128x128x32 block tile, 8 waves (2x4), each wave = 4x2 WMMA tiles.
// Double-buffered LDS (A staged [m][k], B transposed [n][k], stride-40 halfs):
// tile k+1's 8 global_load_b128 are issued before tile k's WMMAs so the
// matrix math hides the HBM latency; one barrier per K-step.
// All tail handling via clamped addresses + selects: no divergent loads,
// EXEC all-ones around every WMMA.
__global__ __launch_bounds__(256) void k_gemm_wmma(
    const float* __restrict__ A, const float* __restrict__ Bm,
    const float* __restrict__ bias, float* __restrict__ C,
    int M, int K, int Nc, int relu) {
  __shared__ __align__(16) _Float16 ldsA[2][BM * LDST];
  __shared__ __align__(16) _Float16 ldsB[2][BN * LDST];

  const int t    = threadIdx.x;
  const int lane = t & 31;
  const int w    = t >> 5;          // wave 0..7
  const int wm   = w >> 2;          // 0..1 : 64-row strip
  const int wn   = w & 3;           // 0..3 : 32-col strip
  const int r    = lane & 15;
  const int hi   = lane >> 4;
  const int kbA  = hi ? 8 : 0;      // A 16x32 f16 layout (ISA 7.12.2)
  const int kbB  = hi ? 8 : 0;      // B 32x16 f16 layout

  const int m0 = blockIdx.y * BM;
  const int n0 = blockIdx.x * BN;

  // per-thread staging coordinates (4 x f4 each for A and B)
  int aRow[4], aCol[4], bRow[4], bCol[4];
#pragma unroll
  for (int i = 0; i < 4; ++i) {
    int q = i * 256 + t;
    aRow[i] = q >> 3;  aCol[i] = (q & 7) << 2;    // A tile BMxBK
    bRow[i] = q >> 5;  bCol[i] = (q & 31) << 2;   // B tile BKxBN
  }

  f4 ta[4], tb[4];   // in-flight global tile (registers)

  // phase 1: issue all 8 global_load_b128 back-to-back (no per-load wait)
  auto globalLoad = [&](int k0) {
#pragma unroll
    for (int i = 0; i < 4; ++i) {
      int kk  = k0 + aCol[i];
      int kcl = kk < K ? kk : (K - 4);            // K % 4 == 0: stays aligned
      ta[i] = *(const f4*)&A[(size_t)(m0 + aRow[i]) * K + kcl];
    }
#pragma unroll
    for (int i = 0; i < 4; ++i) {
      int kk  = k0 + bRow[i];
      int nn  = n0 + bCol[i];
      int kcl = kk < K ? kk : (K - 1);
      int ncl = nn < Nc ? nn : (Nc - 4);          // Nc % 4 == 0
      tb[i] = *(const f4*)&Bm[(size_t)kcl * Nc + ncl];
    }
  };

  // phase 2: convert f32->f16 (zero-fill tails) and store to LDS buffer
  auto ldsStore = [&](int k0, int buf) {
    _Float16* la = ldsA[buf];
    _Float16* lb = ldsB[buf];
#pragma unroll
    for (int i = 0; i < 4; ++i) {
      bool ok = (k0 + aCol[i]) < K;
      h4 hv;
      hv[0] = ok ? (f16)ta[i][0] : (f16)0.f;
      hv[1] = ok ? (f16)ta[i][1] : (f16)0.f;
      hv[2] = ok ? (f16)ta[i][2] : (f16)0.f;
      hv[3] = ok ? (f16)ta[i][3] : (f16)0.f;
      *(h4*)&la[aRow[i] * LDST + aCol[i]] = hv;
    }
#pragma unroll
    for (int i = 0; i < 4; ++i) {
      bool ok = ((k0 + bRow[i]) < K) && ((n0 + bCol[i]) < Nc);
#pragma unroll
      for (int j = 0; j < 4; ++j)                 // transpose: [k][n] -> [n][k]
        lb[(bCol[i] + j) * LDST + bRow[i]] = ok ? (f16)tb[i][j] : (f16)0.f;
    }
  };

  v8f acc[4][2] = {};

  // compute one K-step from LDS buffer: 12 ds_load_b128 + 8 WMMA per wave
  auto compute = [&](int buf) {
    frag16 fa[4], fb[2];
#pragma unroll
    for (int mi = 0; mi < 4; ++mi) {
      const _Float16* p = &ldsA[buf][(wm * 64 + mi * 16 + r) * LDST + kbA];
      fa[mi].h[0] = *(const v8h*)p;
      fa[mi].h[1] = *(const v8h*)(p + 16);
    }
#pragma unroll
    for (int ni = 0; ni < 2; ++ni) {
      const _Float16* p = &ldsB[buf][(wn * 32 + ni * 16 + r) * LDST + kbB];
      fb[ni].h[0] = *(const v8h*)p;
      fb[ni].h[1] = *(const v8h*)(p + 16);
    }
#pragma unroll
    for (int mi = 0; mi < 4; ++mi)
#pragma unroll
      for (int ni = 0; ni < 2; ++ni)
        acc[mi][ni] = __builtin_amdgcn_wmma_f32_16x16x32_f16(
            false, fa[mi].v, false, fb[ni].v, (short)0, acc[mi][ni], false, false);
  };

  const int nk = (K + BK - 1) / BK;
  globalLoad(0);
  ldsStore(0, 0);
  __syncthreads();

  for (int kt = 0; kt < nk; ++kt) {
    const int  curb    = kt & 1;
    const bool hasNext = (kt + 1) < nk;
    if (hasNext) {
      globalLoad((kt + 1) * BK);                  // in flight under the WMMAs
      if (kt + 2 < nk) {                          // warm GL2 two tiles ahead
        __builtin_prefetch(&Bm[(size_t)((kt + 2) * BK) * Nc + n0 + ((t & 31) << 2)], 0, 1);
        __builtin_prefetch(&A[(size_t)(m0 + (t >> 1)) * K + (kt + 2) * BK], 0, 1);
      }
    }
    compute(curb);
    if (hasNext) {
      ldsStore((kt + 1) * BK, curb ^ 1);
      __syncthreads();
    }
  }

  // ---- store (N-edge guard is uniform per 16-col subtile)
#pragma unroll
  for (int mi = 0; mi < 4; ++mi) {
#pragma unroll
    for (int ni = 0; ni < 2; ++ni) {
      int nbase = n0 + wn * 32 + ni * 16;
      if (nbase >= Nc) continue;
      int ncol = nbase + r;
      float bb = bias ? bias[ncol] : 0.f;
#pragma unroll
      for (int rr = 0; rr < 8; ++rr) {
        int mrow = m0 + wm * 64 + mi * 16 + rr + (hi ? 8 : 0);
        float v = acc[mi][ni][rr] + bb;
        if (relu) v = fmaxf(v, 0.f);
        C[(size_t)mrow * Nc + ncol] = v;
      }
    }
  }
}

// ---------------------------------------------------------------- coalesce
__global__ void k_keys_coalesce(const int* __restrict__ cluster,
                                const int* __restrict__ row,
                                const int* __restrict__ col,
                                const int* __restrict__ ev,
                                unsigned long long* __restrict__ key) {
  int e = blockIdx.x * blockDim.x + threadIdx.x;
  if (e >= EE) return;
  const unsigned long long big = (unsigned long long)NN * NN;
  bool live = ev ? (ev[e] != 0) : true;
  if (!live) { key[e] = big; return; }
  int rr = row[e], cc = col[e];
  if (cluster) { rr = cluster[rr]; cc = cluster[cc]; }
  key[e] = (unsigned long long)rr * NN + (unsigned)cc;
}

__global__ void k_bitonic(unsigned long long* __restrict__ key, int j, int k) {
  int i = blockIdx.x * blockDim.x + threadIdx.x;
  int ixj = i ^ j;
  if (ixj <= i || i >= EE) return;
  unsigned long long a = key[i], b = key[ixj];
  bool up = ((i & k) == 0);
  bool swap = up ? (a > b) : (a < b);
  if (swap) { key[i] = b; key[ixj] = a; }
}

__global__ void k_coalesce_extract(const unsigned long long* __restrict__ key,
                                   int* __restrict__ r, int* __restrict__ c,
                                   int* __restrict__ ev) {
  int e = blockIdx.x * blockDim.x + threadIdx.x;
  if (e >= EE) return;
  const unsigned long long big = (unsigned long long)NN * NN;
  unsigned long long kk = key[e];
  bool uniq = (kk < big) && (e == 0 || key[e - 1] != kk);
  r[e]  = uniq ? (int)(kk >> 12) : 0;       // NN == 2^12
  c[e]  = uniq ? (int)(kk & (NN - 1)) : 0;
  ev[e] = uniq ? 1 : 0;
}

// ---------------------------------------------------------------- GCN
__global__ void k_edge_w_deg(const int* __restrict__ r, const int* __restrict__ c,
                             const int* __restrict__ ev, float* __restrict__ wE,
                             float* __restrict__ deg) {
  int e = blockIdx.x * blockDim.x + threadIdx.x;
  if (e >= EE) return;
  float w = (ev[e] && r[e] != c[e]) ? 1.f : 0.f;
  wE[e] = w;
  if (w != 0.f) atomicAdd(&deg[c[e]], w);
}
__global__ void k_dinv(float* deg) {
  int v = blockIdx.x * blockDim.x + threadIdx.x;
  if (v < NN) deg[v] = rsqrtf(deg[v] + 1.f);
}
__global__ void k_gcn_init(const float* __restrict__ h, const float* __restrict__ dinv,
                           const float* __restrict__ b, float* __restrict__ out) {
  int i = blockIdx.x * blockDim.x + threadIdx.x;
  if (i >= NN * FF) return;
  int v = i / FF, f = i - v * FF;
  float di = dinv[v];
  out[i] = h[i] * di * di + b[f];
}
__global__ void k_gcn_scatter(const float* __restrict__ h, const int* __restrict__ r,
                              const int* __restrict__ c, const float* __restrict__ wE,
                              const float* __restrict__ dinv, float* __restrict__ out) {
  int f = blockIdx.x * blockDim.x + threadIdx.x;
  if (f >= FF) return;
  int e = blockIdx.y;
  float w = wE[e];
  if (w == 0.f) return;
  int rr = r[e], cc = c[e];
  float coef = w * dinv[rr] * dinv[cc];
  atomicAdd(&out[(size_t)cc * FF + f], h[(size_t)rr * FF + f] * coef);
}

// ---------------------------------------------------------------- EdgePool
__global__ void k_pool_scores(const float* __restrict__ x, const float* __restrict__ pw,
                              float* __restrict__ s1, float* __restrict__ s2) {
  int v = blockIdx.x * blockDim.x + threadIdx.x;
  if (v >= NN) return;
  const float* xv = x + (size_t)v * FF;
  float a = 0.f, b = 0.f;
  for (int f = 0; f < FF; ++f) { a += xv[f] * pw[f]; b += xv[f] * pw[FF + f]; }
  s1[v] = a; s2[v] = b;
}
__global__ void k_raw_max(const int* __restrict__ r, const int* __restrict__ c,
                          const int* __restrict__ ev, const float* __restrict__ s1,
                          const float* __restrict__ s2, const float* __restrict__ pb,
                          float* __restrict__ rawm, float* __restrict__ mmax) {
  int e = blockIdx.x * blockDim.x + threadIdx.x;
  if (e >= EE) return;
  float raw = s1[r[e]] + s2[c[e]] + pb[0];
  float rm = ev[e] ? raw : -1e30f;
  rawm[e] = rm;
  atomicMaxF(&mmax[c[e]], rm);
}
__global__ void k_pexp(const int* __restrict__ c, const int* __restrict__ ev,
                       const float* __restrict__ rawm, const float* __restrict__ mmax,
                       float* __restrict__ p, float* __restrict__ den) {
  int e = blockIdx.x * blockDim.x + threadIdx.x;
  if (e >= EE) return;
  float z = ev[e] ? rawm[e] - mmax[c[e]] : 0.f;
  float pe = ev[e] ? __expf(z) : 0.f;
  p[e] = pe;
  if (pe != 0.f) atomicAdd(&den[c[e]], pe);
}
__global__ void k_score(const int* __restrict__ c, const float* __restrict__ p,
                        const float* __restrict__ den, float* __restrict__ score) {
  int e = blockIdx.x * blockDim.x + threadIdx.x;
  if (e >= EE) return;
  float d = den[c[e]];
  score[e] = p[e] / (d > 0.f ? d : 1.f) + 0.5f;
}
__global__ void k_sortkeys(const int* __restrict__ ev, const float* __restrict__ score,
                           unsigned long long* __restrict__ key) {
  int e = blockIdx.x * blockDim.x + threadIdx.x;
  if (e >= EE) return;
  float sc = ev[e] ? score[e] : -1.0f;
  unsigned u = __float_as_uint(sc);
  u = (u & 0x80000000u) ? ~u : (u | 0x80000000u);
  key[e] = ((unsigned long long)(~u) << 32) | (unsigned)e;
}
__global__ void k_perm(const unsigned long long* __restrict__ key, int* __restrict__ perm) {
  int e = blockIdx.x * blockDim.x + threadIdx.x;
  if (e < EE) perm[e] = (int)(key[e] & 0xffffffffu);
}

// Greedy max-score matching: inherently sequential (mirrors the fori_loop).
__global__ void k_greedy(const int* __restrict__ perm, const int* __restrict__ r,
                         const int* __restrict__ c, const int* __restrict__ ev,
                         const int* __restrict__ validv, const int* __restrict__ batchv,
                         int* __restrict__ avail, int* __restrict__ cluster,
                         int* __restrict__ chosen, int* __restrict__ newvalid,
                         int* __restrict__ newbatch, int* __restrict__ meta) {
  if (threadIdx.x != 0 || blockIdx.x != 0) return;
  for (int v = 0; v < NN; ++v) { avail[v] = validv[v]; cluster[v] = 0; }
  for (int e = 0; e < EE; ++e) chosen[e] = 0;
  int cnt = 0;
  for (int i = 0; i < EE; ++i) {
    int e = perm[i];
    int s = r[e], d = c[e];
    if (ev[e] && avail[s] && avail[d]) {
      cluster[s] = cnt; cluster[d] = cnt;
      avail[s] = 0; avail[d] = 0;
      chosen[e] = 1; ++cnt;
    }
  }
  int c2 = cnt;
  for (int v = 0; v < NN; ++v) if (avail[v] && validv[v]) cluster[v] = c2++;
  int ncl = c2;
  for (int v = 0; v < NN; ++v) if (!validv[v]) cluster[v] = c2++;
  for (int v = 0; v < NN; ++v) newbatch[v] = NBATCH;
  for (int v = 0; v < NN; ++v) newbatch[cluster[v]] = validv[v] ? batchv[v] : NBATCH;
  for (int v = 0; v < NN; ++v) newvalid[v] = (v < ncl) ? 1 : 0;
  meta[0] = cnt; meta[1] = ncl;
}

__global__ void k_scl_scatter(const int* __restrict__ chosen, const int* __restrict__ cluster,
                              const int* __restrict__ r, const float* __restrict__ score,
                              float* __restrict__ scl) {
  int e = blockIdx.x * blockDim.x + threadIdx.x;
  if (e >= EE) return;
  if (chosen[e]) scl[cluster[r[e]]] = score[e];
}
__global__ void k_pool_sum(const float* __restrict__ x, const int* __restrict__ cluster,
                           float* __restrict__ newx) {
  int f = blockIdx.x * blockDim.x + threadIdx.x;
  if (f >= FF) return;
  int v = blockIdx.y;
  atomicAdd(&newx[(size_t)cluster[v] * FF + f], x[(size_t)v * FF + f]);
}
__global__ void k_scale_relu(float* __restrict__ newx, const float* __restrict__ scl) {
  int i = blockIdx.x * blockDim.x + threadIdx.x;
  if (i >= NN * FF) return;
  int cl = i / FF;
  newx[i] = fmaxf(newx[i] * scl[cl], 0.f);
}

// ---------------------------------------------------------------- readout
__global__ void k_gmax(const float* __restrict__ x, const int* __restrict__ validv,
                       const int* __restrict__ batchv, float* __restrict__ g) {
  int i = blockIdx.x * blockDim.x + threadIdx.x;
  if (i >= NN * FF) return;
  int v = i / FF, f = i - v * FF;
  float val = validv[v] ? x[i] : -1e30f;
  int b = validv[v] ? batchv[v] : 0;
  atomicMaxF(&g[(size_t)b * FF + f], val);
}
__global__ void k_linear_relu(const float* __restrict__ A, const float* __restrict__ W,
                              const float* __restrict__ b, float* __restrict__ out,
                              int M, int K, int Nc) {
  int i = blockIdx.x * blockDim.x + threadIdx.x;
  if (i >= M * Nc) return;
  int m = i / Nc, n = i - m * Nc;
  float acc = b[n];
  for (int k = 0; k < K; ++k) acc += A[(size_t)m * K + k] * W[(size_t)k * Nc + n];
  out[i] = fmaxf(acc, 0.f);
}

// ---------------------------------------------------------------- host
static inline char* wsAlloc(char*& cur, size_t bytes) {
  char* p = cur;
  cur += (bytes + 255) & ~(size_t)255;
  return p;
}

extern "C" void kernel_launch(void* const* d_in, const int* in_sizes, int n_in,
                              void* d_out, int out_size, void* d_ws, size_t ws_size,
                              hipStream_t stream) {
  (void)in_sizes; (void)n_in; (void)out_size; (void)ws_size;
  const int*   tokens  = (const int*)d_in[0];
  const int*   rowIn   = (const int*)d_in[1];
  const int*   colIn   = (const int*)d_in[2];
  const int*   batchIn = (const int*)d_in[3];
  const float* emb = (const float*)d_in[4];
  const float* Ws[3]  = {(const float*)d_in[5],  (const float*)d_in[9],  (const float*)d_in[13]};
  const float* bs[3]  = {(const float*)d_in[6],  (const float*)d_in[10], (const float*)d_in[14]};
  const float* pws[3] = {(const float*)d_in[7],  (const float*)d_in[11], (const float*)d_in[15]};
  const float* pbs[3] = {(const float*)d_in[8],  (const float*)d_in[12], (const float*)d_in[16]};
  const float* L1 = (const float*)d_in[17]; const float* bl1 = (const float*)d_in[18];
  const float* L2 = (const float*)d_in[19]; const float* bl2 = (const float*)d_in[20];
  const float* L3 = (const float*)d_in[21]; const float* bl3 = (const float*)d_in[22];
  const int Ks[3] = {FIN, FF, FF};

  char* cur = (char*)d_ws;
  float* bufX = (float*)wsAlloc(cur, (size_t)NN * FF * 4);
  float* bufH = (float*)wsAlloc(cur, (size_t)NN * FF * 4);
  float* bufO = (float*)wsAlloc(cur, (size_t)NN * FF * 4);
  float* deg  = (float*)wsAlloc(cur, NN * 4);
  float* wE   = (float*)wsAlloc(cur, EE * 4);
  unsigned long long* keys = (unsigned long long*)wsAlloc(cur, EE * 8);
  int* rA  = (int*)wsAlloc(cur, EE * 4); int* cA  = (int*)wsAlloc(cur, EE * 4);
  int* evA = (int*)wsAlloc(cur, EE * 4);
  int* rB  = (int*)wsAlloc(cur, EE * 4); int* cB  = (int*)wsAlloc(cur, EE * 4);
  int* evB = (int*)wsAlloc(cur, EE * 4);
  int* permv  = (int*)wsAlloc(cur, EE * 4);
  int* chosen = (int*)wsAlloc(cur, EE * 4);
  float* rawm  = (float*)wsAlloc(cur, EE * 4);
  float* pE    = (float*)wsAlloc(cur, EE * 4);
  float* score = (float*)wsAlloc(cur, EE * 4);
  float* s1 = (float*)wsAlloc(cur, NN * 4);
  float* s2 = (float*)wsAlloc(cur, NN * 4);
  float* mmax = (float*)wsAlloc(cur, NN * 4);
  float* den  = (float*)wsAlloc(cur, NN * 4);
  int* validA = (int*)wsAlloc(cur, NN * 4); int* validB = (int*)wsAlloc(cur, NN * 4);
  int* batchA = (int*)wsAlloc(cur, NN * 4); int* batchB = (int*)wsAlloc(cur, NN * 4);
  int* avail   = (int*)wsAlloc(cur, NN * 4);
  int* cluster = (int*)wsAlloc(cur, NN * 4);
  int* meta    = (int*)wsAlloc(cur, 64);
  float* scl = (float*)wsAlloc(cur, (NN + 1) * 4);
  float* g   = (float*)wsAlloc(cur, (size_t)NBATCH * FF * 4);
  float* h1  = (float*)wsAlloc(cur, NBATCH * 1000 * 4);
  float* h2  = (float*)wsAlloc(cur, NBATCH * 500 * 4);

  const int T = 256;
  const int gE  = (EE + T - 1) / T;
  const int gN  = (NN + T - 1) / T;
  const int gNF = (int)(((size_t)NN * FF + T - 1) / T);
  const int gF  = (FF + T - 1) / T;

  auto bitonic = [&](unsigned long long* key) {
    for (int k = 2; k <= EE; k <<= 1)
      for (int j = k >> 1; j > 0; j >>= 1)
        k_bitonic<<<gE, T, 0, stream>>>(key, j, k);
  };

  // x = emb[tokens].reshape(NN, FIN)
  k_embed<<<(NN * FIN + T - 1) / T, T, 0, stream>>>(tokens, emb, bufX);
  k_fill_i32<<<gN, T, 0, stream>>>(validA, 1, NN);
  k_copy_i32<<<gN, T, 0, stream>>>(batchA, batchIn, NN);
  // initial coalesce
  k_keys_coalesce<<<gE, T, 0, stream>>>(nullptr, rowIn, colIn, nullptr, keys);
  bitonic(keys);
  k_coalesce_extract<<<gE, T, 0, stream>>>(keys, rA, cA, evA);

  float* x = bufX;
  int *rc = rA, *cc = cA, *evc = evA;
  int *rn = rB, *cn = cB, *evn = evB;
  int *validc = validA, *validn = validB;
  int *batchc = batchA, *batchn = batchB;

  for (int l = 0; l < 3; ++l) {
    // ---- GCN layer l
    k_fill_f32<<<gN, T, 0, stream>>>(deg, 0.f, NN);
    k_edge_w_deg<<<gE, T, 0, stream>>>(rc, cc, evc, wE, deg);
    k_dinv<<<gN, T, 0, stream>>>(deg);                 // deg -> dinv in place
    k_gemm_wmma<<<dim3((FF + BN - 1) / BN, NN / BM), 256, 0, stream>>>(
        x, Ws[l], nullptr, bufH, NN, Ks[l], FF, 0);    // h = x @ W
    k_gcn_init<<<gNF, T, 0, stream>>>(bufH, deg, bs[l], bufO);
    k_gcn_scatter<<<dim3(gF, EE), T, 0, stream>>>(bufH, rc, cc, wE, deg, bufO);
    // ---- EdgePool layer l  (input features in bufO)
    k_pool_scores<<<gN, T, 0, stream>>>(bufO, pws[l], s1, s2);
    k_fill_f32<<<gN, T, 0, stream>>>(mmax, -1e30f, NN);
    k_raw_max<<<gE, T, 0, stream>>>(rc, cc, evc, s1, s2, pbs[l], rawm, mmax);
    k_fill_f32<<<gN, T, 0, stream>>>(den, 0.f, NN);
    k_pexp<<<gE, T, 0, stream>>>(cc, evc, rawm, mmax, pE, den);
    k_score<<<gE, T, 0, stream>>>(cc, pE, den, score);
    k_sortkeys<<<gE, T, 0, stream>>>(evc, score, keys);
    bitonic(keys);
    k_perm<<<gE, T, 0, stream>>>(keys, permv);
    k_greedy<<<1, 1, 0, stream>>>(permv, rc, cc, evc, validc, batchc,
                                  avail, cluster, chosen, validn, batchn, meta);
    k_fill_f32<<<(NN + 1 + T - 1) / T, T, 0, stream>>>(scl, 1.f, NN + 1);
    k_scl_scatter<<<gE, T, 0, stream>>>(chosen, cluster, rc, score, scl);
    k_fill_f32<<<gNF, T, 0, stream>>>(x, 0.f, NN * FF); // newx into old x buffer
    k_pool_sum<<<dim3(gF, NN), T, 0, stream>>>(bufO, cluster, x);
    k_scale_relu<<<gNF, T, 0, stream>>>(x, scl);        // x = relu(scl * segsum)
    // coalesce remapped edges
    k_keys_coalesce<<<gE, T, 0, stream>>>(cluster, rc, cc, evc, keys);
    bitonic(keys);
    k_coalesce_extract<<<gE, T, 0, stream>>>(keys, rn, cn, evn);
    // swap edge / valid / batch buffers
    int* t;
    t = rc; rc = rn; rn = t;  t = cc; cc = cn; cn = t;  t = evc; evc = evn; evn = t;
    t = validc; validc = validn; validn = t;
    t = batchc; batchc = batchn; batchn = t;
  }

  // global max-pool + MLP head
  k_fill_f32<<<(NBATCH * FF + T - 1) / T, T, 0, stream>>>(g, -3.0e38f, NBATCH * FF);
  k_gmax<<<gNF, T, 0, stream>>>(x, validc, batchc, g);
  k_linear_relu<<<(NBATCH * 1000 + T - 1) / T, T, 0, stream>>>(g,  L1, bl1, h1, NBATCH, FF,   1000);
  k_linear_relu<<<(NBATCH * 500  + T - 1) / T, T, 0, stream>>>(h1, L2, bl2, h2, NBATCH, 1000, 500);
  k_linear_relu<<<1, 32, 0, stream>>>(h2, L3, bl3, (float*)d_out, NBATCH, 500, 4);
}
